// RNN_MultiRegional_D1_58420145160255
// MI455X (gfx1250) — matrix operand order repaired and checked by reading, not compiled, not run on hardware
//
#include <hip/hip_runtime.h>
#include <hip/hip_bf16.h>

// ---------------- problem constants ----------------
#define HID     512
#define FOURH   2048
#define NB      32            // batch
#define NT      1000          // time steps
#define DT      0.01f
#define HKEEP   359           // H - int(0.3*H) = 512 - 153
// ---------------- launch geometry ------------------
#define NWG         64
#define COLS_PER_WG 32        // FOURH / NWG
#define LDS_STRIDE  2056      // 2048 + 8 bf16 elems -> 4112B row stride (16B aligned, bank spread)

typedef __bf16 bf16;
typedef __attribute__((ext_vector_type(16))) __bf16 v16bf;
typedef __attribute__((ext_vector_type(8)))  __bf16 v8bf;
typedef __attribute__((ext_vector_type(8)))  float  v8f;
typedef __attribute__((ext_vector_type(4)))  float  f32x4;
typedef __attribute__((ext_vector_type(4)))  unsigned int u32x4;

union V16 { u32x4 q[2]; v16bf v; };

__device__ __forceinline__ float ht(float x) {
    x = x < 1e-15f ? 1e-15f : x;
    return x > 1.0f ? 1.0f : x;
}

// sense/generation grid barrier in device memory (bar[0]=count, bar[1]=generation)
__device__ __forceinline__ void grid_barrier(int* bar) {
    __syncthreads();
    if (threadIdx.x == 0) {
        __threadfence();
        int* cnt = bar;
        int* gen = bar + 1;
        int g = __hip_atomic_load(gen, __ATOMIC_RELAXED, __HIP_MEMORY_SCOPE_AGENT);
        int arrived = __hip_atomic_fetch_add(cnt, 1, __ATOMIC_ACQ_REL, __HIP_MEMORY_SCOPE_AGENT);
        if (arrived == NWG - 1) {
            __hip_atomic_store(cnt, 0, __ATOMIC_RELAXED, __HIP_MEMORY_SCOPE_AGENT);
            __hip_atomic_fetch_add(gen, 1, __ATOMIC_RELEASE, __HIP_MEMORY_SCOPE_AGENT);
        } else {
            while (__hip_atomic_load(gen, __ATOMIC_ACQUIRE, __HIP_MEMORY_SCOPE_AGENT) == g) {
                __builtin_amdgcn_s_sleep(2);
            }
        }
    }
    __syncthreads();
}

// seed bf16 h ping buffer 0 from hn, zero barrier state (re-run each call -> deterministic)
__global__ void rnn_init_kernel(const float* __restrict__ hn,
                                bf16* __restrict__ h16,
                                int* __restrict__ bar) {
    int i = blockIdx.x * blockDim.x + threadIdx.x;
    if (i < NB * FOURH) h16[i] = (bf16)hn[i];
    if (i == 0) { bar[0] = 0; bar[1] = 0; }
}

__global__ __launch_bounds__(128, 1)
void rnn_persistent_kernel(const float* __restrict__ inp,     // [B,T,1]
                           const float* __restrict__ hn,      // [1,B,4H]
                           const float* __restrict__ inhib,   // [B,T,4H]
                           const float* __restrict__ s2s_fix, // [H,H]
                           const float* __restrict__ t2a_w,
                           const float* __restrict__ t2s_w,
                           const float* __restrict__ a2a_w,
                           const float* __restrict__ a2s_w,
                           const float* __restrict__ s2n_w,
                           const float* __restrict__ n2t_w,
                           const float* __restrict__ iw,      // [1,4H]
                           bf16*  __restrict__ h16,           // [2][B,4H] ping-pong matmul operand
                           int*   __restrict__ bar,
                           float* __restrict__ out)           // [B*4H hn_last | B*T*4H rnn_out]
{
    __shared__ bf16 Wl[COLS_PER_WG * LDS_STRIDE];   // 131584 B of this WGP's 320 KB

    const int tid = threadIdx.x;
    const int n0  = blockIdx.x * COLS_PER_WG;       // global W_rec row (output neuron) base

    // ---- one-time: build constrained W_rec rows [n0, n0+32) in bf16 into LDS ----
    for (int e = tid; e < COLS_PER_WG * FOURH; e += 128) {
        int jl = e >> 11;                // local row 0..31  (output neuron)
        int k  = e & (FOURH - 1);        // input index
        int j  = n0 + jl;
        int jb = j >> 9, kb = k >> 9;    // 512-blocks
        int jj = j & 511, kk = k & 511;
        int idx = jj * 512 + kk;
        float w = 0.0f;
        if (jb == 0) {                                    // striatum rows
            if      (kb == 0) w = -s2s_fix[idx];                          // -(0*ht + fixed)
            else if (kb == 2) w = (jj < 256) ? 0.0f : ht(t2s_w[idx]);     // thal2str row mask
            else if (kb == 3) w = (kk < HKEEP) ? ht(a2s_w[idx]) : 0.0f;   // alm2str col mask
        } else if (jb == 1) {                             // SNr rows
            if (kb == 0) w = -ht(s2n_w[idx]);
        } else if (jb == 2) {                             // thal rows
            if (kb == 1) w = -ht(n2t_w[idx]);
        } else {                                          // ALM rows
            if      (kb == 2) w = ht(t2a_w[idx]);
            else if (kb == 3) w = ht(a2a_w[idx]) * ((kk < HKEEP) ? 1.0f : -1.0f);
        }
        Wl[jl * LDS_STRIDE + k] = (bf16)w;
    }
    __syncthreads();

    // ---- wave / tile assignment: A = W slice (M = neuron), B = h (N = batch) ----
    const int wave  = tid >> 5;           // 4 waves
    const int lane  = tid & 31;
    const int mt    = wave & 1;           // neuron tile within slice
    const int btile = wave >> 1;          // batch tile
    const int g     = lane >> 4;          // lane group (ISA A/B/C layout)
    const int ln    = lane & 15;
    const int b     = btile * 16 + ln;    // this lane's batch row (C/D column N)
    const int m0    = mt * 16;            // local neuron tile base
    const int aml   = m0 + ln;            // A-operand (W) local row for this lane
    const int jbase = n0 + m0 + 8 * g;    // lane owns 8 consecutive neurons j = jbase..jbase+7

    // per-neuron drive constants for the 8 owned columns
    float iwm[8], ton[8];
    #pragma unroll
    for (int r = 0; r < 8; ++r) {
        int j = jbase + r;
        iwm[r] = ht(iw[j]) * ((j >= 256 && j < 512) ? 1.0f : 0.0f);   // strthal mask
        ton[r] = (j >= 512 && j < 1024) ? 0.7f : ((j >= 1024 && j < 1536) ? 1.0f : 0.0f);
    }

    // persistent fp32 state h[b][jbase+r] lives in registers for the whole sequence
    float hreg[8];
    {
        f32x4 h0 = *(const f32x4*)(hn + (size_t)b * FOURH + jbase);
        f32x4 h1 = *(const f32x4*)(hn + (size_t)b * FOURH + jbase + 4);
        #pragma unroll
        for (int r = 0; r < 4; ++r) { hreg[r] = h0[r]; hreg[r + 4] = h1[r]; }
    }

    const float* inpPtr = inp + (size_t)b * NT;                              // [t]
    const float* inhRow = inhib + ((size_t)b * NT) * FOURH + jbase;          // += 4H per t
    float*       outRow = out + (size_t)NB * FOURH
                              + ((size_t)b * NT) * FOURH + jbase;            // += 4H per t

    v8f hv = {0.f, 0.f, 0.f, 0.f, 0.f, 0.f, 0.f, 0.f};

    for (int t = 0; t < NT; ++t) {
        const bf16* hcur  = h16 + (size_t)( t      & 1) * (NB * FOURH);
        bf16*       hnext = h16 + (size_t)((t + 1) & 1) * (NB * FOURH);

        // issue this step's drive loads up front: latency hides behind 64 WMMAs
        float ip = inpPtr[t];
        f32x4 i0 = *(const f32x4*)(inhRow);
        f32x4 i1 = *(const f32x4*)(inhRow + 4);
        __builtin_prefetch(inhRow + FOURH, 0, 1);   // next step's inhib line (speculative)

        // ---- out[j,b] = sum_k W[j,k] * h[b,k] ----
        v8f acc = {0.f, 0.f, 0.f, 0.f, 0.f, 0.f, 0.f, 0.f};
        const bf16* brow = hcur + (size_t)b * FOURH;
        #pragma unroll 4
        for (int kb = 0; kb < FOURH; kb += 32) {
            V16 a, bb;
            // 16-bit A layout: v0..3 = K g*8+[0..7], v4..7 = K 16+g*8+[0..7]
            a.q[0]  = *(const u32x4*)&Wl[aml * LDS_STRIDE + kb + g * 8];
            a.q[1]  = *(const u32x4*)&Wl[aml * LDS_STRIDE + kb + 16 + g * 8];
            // 16-bit B layout: v0..3 = K g*8+[0..7], v4..7 = K 16+g*8+[0..7], col n = batch
            bb.q[0] = *(const u32x4*)(brow + kb + g * 8);
            bb.q[1] = *(const u32x4*)(brow + kb + 16 + g * 8);
            acc = __builtin_amdgcn_wmma_f32_16x16x32_bf16(
                      false, a.v, false, bb.v, (short)0, acc, false, false);
        }

        // ---- fused decay + drive + relu; acc[r] -> (j = jbase+r, b) : all contiguous ----
        v8bf hb;
        #pragma unroll
        for (int r = 0; r < 8; ++r) {
            float inh   = (r < 4) ? i0[r] : i1[r - 4];
            float drive = ip * iwm[r] + inh + ton[r];
            float hnew  = (1.0f - DT) * hreg[r] + DT * (acc[r] + drive);
            hnew = hnew > 0.0f ? hnew : 0.0f;
            hreg[r] = hnew;
            hv[r]   = hnew;
            hb[r]   = (bf16)hnew;
        }
        *(v8f*)outRow = hv;                                        // rnn_out[b,t,jbase..+7]
        *(v8bf*)(hnext + (size_t)b * FOURH + jbase) = hb;          // next matmul operand

        inhRow += FOURH;
        outRow += FOURH;

        grid_barrier(bar);   // publish hnext before anyone reads it at t+1
    }

    // hn_last[b, jbase..+7] from live registers (no per-step branch)
    *(v8f*)(out + (size_t)b * FOURH + jbase) = hv;
}

extern "C" void kernel_launch(void* const* d_in, const int* in_sizes, int n_in,
                              void* d_out, int out_size, void* d_ws, size_t ws_size,
                              hipStream_t stream) {
    const float* inp    = (const float*)d_in[0];
    const float* hn     = (const float*)d_in[1];
    const float* inhib  = (const float*)d_in[2];
    // d_in[3] = str2str_w: multiplied by 0 in reference, unused
    const float* t2a_w  = (const float*)d_in[4];
    const float* t2s_w  = (const float*)d_in[5];
    const float* a2a_w  = (const float*)d_in[6];
    const float* a2s_w  = (const float*)d_in[7];
    const float* s2n_w  = (const float*)d_in[8];
    const float* n2t_w  = (const float*)d_in[9];
    const float* iw     = (const float*)d_in[10];
    const float* s2sfix = (const float*)d_in[11];
    float* out = (float*)d_out;

    // workspace layout: bf16 h ping-pong (256KB) | barrier (8B)
    char* ws   = (char*)d_ws;
    bf16* h16  = (bf16*)ws;
    int*  bar  = (int*)(ws + (size_t)2 * NB * FOURH * sizeof(bf16));

    rnn_init_kernel<<<(NB * FOURH + 255) / 256, 256, 0, stream>>>(hn, h16, bar);
    rnn_persistent_kernel<<<NWG, 128, 0, stream>>>(
        inp, hn, inhib, s2sfix, t2a_w, t2s_w, a2a_w, a2s_w, s2n_w, n2t_w, iw,
        h16, bar, out);
}
